// DistanceClassifier_13881334300951
// MI455X (gfx1250) — matrix operand order repaired and checked by reading, not compiled, not run on hardware
//
#include <hip/hip_runtime.h>
#include <hip/hip_bf16.h>
#include <math.h>

typedef __attribute__((ext_vector_type(16))) __bf16        v16bf;
typedef __attribute__((ext_vector_type(8)))  float         v8f;
typedef __attribute__((ext_vector_type(8)))  unsigned int  v8u;

#define N_CTX 1024
#define M_RES 256
#define DIM   1024

// ---------------------------------------------------------------------------
// Kernel 0: row squared-norms.  One wave32 per row; rows [0,1024) are context
// rows, rows [1024,1280) are result rows.  160 blocks * 256 thr = 1280 waves.
// ---------------------------------------------------------------------------
__global__ __launch_bounds__(256) void norms_kernel(const float* __restrict__ C,
                                                    const float* __restrict__ R,
                                                    float* __restrict__ c_sq,
                                                    float* __restrict__ r_sq) {
    const int gwave = (blockIdx.x * 256 + threadIdx.x) >> 5;
    const int lane  = threadIdx.x & 31;
    if (gwave >= N_CTX + M_RES) return;

    const float* src = (gwave < N_CTX) ? (C + (size_t)gwave * DIM)
                                       : (R + (size_t)(gwave - N_CTX) * DIM);
    float s = 0.0f;
#pragma unroll
    for (int i = 0; i < DIM; i += 128) {            // 32 lanes * float4 = 128/pass
        float4 v = *(const float4*)(src + i + lane * 4);
        s += v.x * v.x + v.y * v.y + v.z * v.z + v.w * v.w;
    }
#pragma unroll
    for (int off = 16; off >= 1; off >>= 1)
        s += __shfl_xor(s, off, 32);
    if (lane == 0) {
        if (gwave < N_CTX) c_sq[gwave] = s;
        else               r_sq[gwave - N_CTX] = s;
    }
}

// ---------------------------------------------------------------------------
// Kernel 1: fused  cross-GEMM (bf16 WMMA) -> L2 distance -> softmax -> 1-p.
// One workgroup (8 wave32) per 16-row strip; each wave owns two 16-col tiles
// (8 waves * 32 cols = 256 cols = full softmax row in one workgroup).
// ---------------------------------------------------------------------------
__global__ __launch_bounds__(256) void fused_dist_softmax_kernel(
        const float* __restrict__ C,     // [1024,1024] contexts
        const float* __restrict__ R,     // [256,1024]  results
        const float* __restrict__ c_sq,  // [1024]
        const float* __restrict__ r_sq,  // [256]
        float* __restrict__ out)         // [1024,256]
{
    __shared__ __align__(16) __bf16 Abf[16][32];   // staged A strip (1 KB)
    __shared__ float dist[16][257];                // distances, padded rows
    __shared__ float red[16][17];                  // softmax reduction scratch
    __shared__ float rowMax[16];
    __shared__ float rowSum[16];

    const int tid  = threadIdx.x;
    const int lane = tid & 31;
    const int wave = tid >> 5;          // 0..7
    const int lrow = lane & 15;         // 0..15
    const int half = lane >> 4;         // 0 or 1
    const int n0   = blockIdx.x * 16;   // row strip base

    const int col0 = wave * 32;         // this wave's first 16-col tile
    const int col1 = wave * 32 + 16;    // second tile

    v8f acc0 = {};
    v8f acc1 = {};

    for (int k0 = 0; k0 < DIM; k0 += 32) {
        // ---- cooperative stage of A strip (16 rows x 32 K, f32 -> bf16) ----
        {
            const int arow = tid >> 4;           // 0..15
            const int acol = (tid & 15) * 2;     // 0,2,...,30
            const float2 f = *(const float2*)(C + (size_t)(n0 + arow) * DIM + k0 + acol);
            Abf[arow][acol]     = (__bf16)f.x;
            Abf[arow][acol + 1] = (__bf16)f.y;
        }
        __syncthreads();

        // ---- A fragment (contiguous K layout: lanes 0-15 K0..15,
        //      lanes 16-31 K16..31; 2 K per VGPR) ----
        v8u araw;
        {
            uint4 q0 = *(const uint4*)(&Abf[lrow][half * 16]);
            uint4 q1 = *(const uint4*)(&Abf[lrow][half * 16 + 8]);
            araw[0] = q0.x; araw[1] = q0.y; araw[2] = q0.z; araw[3] = q0.w;
            araw[4] = q1.x; araw[5] = q1.y; araw[6] = q1.z; araw[7] = q1.w;
        }
        v16bf a = __builtin_bit_cast(v16bf, araw);

        // ---- B fragments straight from R (row-major => K-run contiguous).
        //      Contiguous K layout, same as A: lanes 0-15 hold K0..15, lanes 16-31 hold K16..31.
        v16bf b0, b1;
        {
            const float4* pb0 = (const float4*)(R + (size_t)(col0 + lrow) * DIM + k0 + half * 16);
            const float4* pb1 = (const float4*)(R + (size_t)(col1 + lrow) * DIM + k0 + half * 16);
#pragma unroll
            for (int q = 0; q < 4; ++q) {
                float4 f0 = pb0[q];
                float4 f1 = pb1[q];
                b0[q * 4 + 0] = (__bf16)f0.x; b0[q * 4 + 1] = (__bf16)f0.y;
                b0[q * 4 + 2] = (__bf16)f0.z; b0[q * 4 + 3] = (__bf16)f0.w;
                b1[q * 4 + 0] = (__bf16)f1.x; b1[q * 4 + 1] = (__bf16)f1.y;
                b1[q * 4 + 2] = (__bf16)f1.z; b1[q * 4 + 3] = (__bf16)f1.w;
            }
        }

        // D = A x B + C   (f32 accumulate)
        acc0 = __builtin_amdgcn_wmma_f32_16x16x32_bf16(false, a, false, b0,
                                                       (short)0, acc0, false, false);
        acc1 = __builtin_amdgcn_wmma_f32_16x16x32_bf16(false, a, false, b1,
                                                       (short)0, acc1, false, false);

        __syncthreads();   // protect Abf before next iteration's store
    }

    // ---- epilogue: squared distance -> sqrt -> stash in LDS ----
    {
        const float rs0 = r_sq[col0 + lrow];
        const float rs1 = r_sq[col1 + lrow];
#pragma unroll
        for (int i = 0; i < 8; ++i) {
            const int rl = half * 8 + i;           // local row (C/D layout)
            const float cs = c_sq[n0 + rl];
            float s0 = fmaxf(cs - 2.0f * acc0[i] + rs0, 0.0f);
            float s1 = fmaxf(cs - 2.0f * acc1[i] + rs1, 0.0f);
            dist[rl][col0 + lrow] = sqrtf(s0);
            dist[rl][col1 + lrow] = sqrtf(s1);
        }
    }
    __syncthreads();

    // ---- softmax over 256 columns per row, then probabilities = 1 - p ----
    const int row   = tid >> 4;    // 0..15
    const int chunk = tid & 15;    // 0..15 (16 cols each)

    float lmax = -INFINITY;
#pragma unroll
    for (int j = 0; j < 16; ++j)
        lmax = fmaxf(lmax, dist[row][chunk * 16 + j]);
    red[row][chunk] = lmax;
    __syncthreads();
    if (tid < 16) {
        float m = red[tid][0];
#pragma unroll
        for (int j = 1; j < 16; ++j) m = fmaxf(m, red[tid][j]);
        rowMax[tid] = m;
    }
    __syncthreads();

    const float rm = rowMax[row];
    float lsum = 0.0f;
#pragma unroll
    for (int j = 0; j < 16; ++j)
        lsum += __expf(dist[row][chunk * 16 + j] - rm);
    red[row][chunk] = lsum;
    __syncthreads();
    if (tid < 16) {
        float s = 0.0f;
#pragma unroll
        for (int j = 0; j < 16; ++j) s += red[tid][j];
        rowSum[tid] = s;
    }
    __syncthreads();

    const float inv = 1.0f / rowSum[row];
    float* op = out + (size_t)(n0 + row) * M_RES + chunk * 16;
#pragma unroll
    for (int j = 0; j < 16; ++j)
        op[j] = 1.0f - __expf(dist[row][chunk * 16 + j] - rm) * inv;
}

// ---------------------------------------------------------------------------
extern "C" void kernel_launch(void* const* d_in, const int* in_sizes, int n_in,
                              void* d_out, int out_size, void* d_ws, size_t ws_size,
                              hipStream_t stream) {
    const float* C = (const float*)d_in[0];   // [1024,1024] f32
    const float* R = (const float*)d_in[1];   // [256,1024]  f32
    float* out  = (float*)d_out;              // [1024,256]  f32
    float* c_sq = (float*)d_ws;               // 1024 f32
    float* r_sq = c_sq + N_CTX;               // 256 f32  (5 KB total scratch)

    norms_kernel<<<(N_CTX + M_RES) * 32 / 256, 256, 0, stream>>>(C, R, c_sq, r_sq);
    fused_dist_softmax_kernel<<<N_CTX / 16, 256, 0, stream>>>(C, R, c_sq, r_sq, out);
}
